// Geoconv_472446403135
// MI455X (gfx1250) — matrix operand order repaired and checked by reading, not compiled, not run on hardware
//
#include <hip/hip_runtime.h>
#include <hip/hip_bf16.h>
#include <hip/hip_fp16.h>

typedef __attribute__((ext_vector_type(16))) _Float16 v16h;
typedef __attribute__((ext_vector_type(8)))  _Float16 v8h;
typedef __attribute__((ext_vector_type(8)))  float    v8f;

#define RADIUS2   (0.15f*0.15f)
#define DRADIUS2  (0.30f*0.30f)
#define BN_EPS    1e-5f
#define NROWS     32768          // B*N
#define NPTS      2048
#define NB        16
#define SROW      40             // LDS row stride in halves (80B: 16B-aligned, bank-friendly)

static __device__ __forceinline__ v8f wmma_f16(v16h a, v16h b, v8f c) {
    return __builtin_amdgcn_wmma_f32_16x16x32_f16(false, a, false, b, (short)0, c, false, false);
}

// load 16 contiguous f16 (32B) as two b128 loads (works for global or LDS pointers)
static __device__ __forceinline__ v16h load_v16h(const _Float16* p) {
    v8h lo = *(const v8h*)p;
    v8h hh = *(const v8h*)(p + 8);
    v16h r;
#pragma unroll
    for (int i = 0; i < 8; ++i) { r[i] = lo[i]; r[i + 8] = hh[i]; }
    return r;
}

// load 8 contiguous f32 (two b128) and convert into halves [base..base+7]
static __device__ __forceinline__ void cvt8(const float* p, v16h& d, int base) {
    float4 a = *(const float4*)p;
    float4 b = *(const float4*)(p + 4);
    d[base + 0] = (_Float16)a.x; d[base + 1] = (_Float16)a.y;
    d[base + 2] = (_Float16)a.z; d[base + 3] = (_Float16)a.w;
    d[base + 4] = (_Float16)b.x; d[base + 5] = (_Float16)b.y;
    d[base + 6] = (_Float16)b.z; d[base + 7] = (_Float16)b.w;
}

// ---------------------------------------------------------------------------
// K1: fused input GEMMs. cols 0:64 = feat@W_feat+b_feat, cols 64:256 = feat@W_byp.
// One wave per n-tile, looping 16 m-tiles; K=64 -> two chained f16 WMMAs.
// ntile is forced scalar (readfirstlane) so the store branch is wave-uniform.
// ---------------------------------------------------------------------------
__global__ void __launch_bounds__(128) k_gemm_in(
    const float* __restrict__ feat, const float* __restrict__ W_feat,
    const float* __restrict__ b_feat, const float* __restrict__ W_byp,
    float* __restrict__ self_feat, float* __restrict__ mutual_raw)
{
    const int lane = threadIdx.x & 31;
    const int hi   = lane >> 4;
    const int col  = lane & 15;
    const int ntile = __builtin_amdgcn_readfirstlane(blockIdx.y * 4 + (threadIdx.x >> 5));
    const bool is_self = (ntile < 4);          // wave-uniform
    const int n = ntile * 16 + col;            // global output column 0..255

    const float* wsrc = is_self ? W_feat : W_byp;
    const int wstride = is_self ? 64 : 192;
    const int wcol    = is_self ? n : (n - 64);

    // B tiles (weights), loaded once per wave, scattered f32 -> f16
    v16h b0, b1;
#pragma unroll
    for (int h = 0; h < 16; ++h) {
        int k = (h & 7) + hi * 8 + ((h >= 8) ? 16 : 0);   // B layout: K = (h&7) + 8*hi + (h>=8?16:0)
        b0[h] = (_Float16)wsrc[k * wstride + wcol];
        b1[h] = (_Float16)wsrc[(k + 32) * wstride + wcol];
    }
    float bias = is_self ? b_feat[n] : 0.0f;

    for (int t = 0; t < 16; ++t) {
        int mbase = (blockIdx.x * 16 + t) * 16;
        int row = mbase + col;                  // A layout: M = lane&15
        const float* fr = feat + (size_t)row * 64;
        v16h a0, a1;                            // A layout: K = h + 8*hi + (h>=8?8:0)
        cvt8(fr + hi * 8,        a0, 0);
        cvt8(fr + 16 + hi * 8,   a0, 8);
        cvt8(fr + 32 + hi * 8,   a1, 0);
        cvt8(fr + 48 + hi * 8,   a1, 8);
        v8f c = {};
        c = wmma_f16(a0, b0, c);
        c = wmma_f16(a1, b1, c);
        if (is_self) {                          // uniform branch, no exec juggling
#pragma unroll
            for (int r = 0; r < 8; ++r) {
                int orow = mbase + r + hi * 8;  // C layout: M = r + 8*hi, N = lane&15
                self_feat[(size_t)orow * 64 + n] = c[r] + bias;
            }
        } else {
#pragma unroll
            for (int r = 0; r < 8; ++r) {
                int orow = mbase + r + hi * 8;
                mutual_raw[(size_t)orow * 192 + (n - 64)] = c[r];
            }
        }
    }
}

// ---------------------------------------------------------------------------
// K2: per-channel sum / sumsq reduction for training-mode BN.
// ---------------------------------------------------------------------------
__global__ void k_stats(const float* __restrict__ x, float* __restrict__ out,
                        int C, int rows_per_block)
{
    int c = threadIdx.x;
    int r0 = blockIdx.x * rows_per_block;
    float s = 0.f, q = 0.f;
    for (int r = 0; r < rows_per_block; ++r) {
        float v = x[(size_t)(r0 + r) * C + c];
        s += v; q += v * v;
    }
    atomicAdd(&out[c], s);
    atomicAdd(&out[C + c], q);
}

// ---------------------------------------------------------------------------
// K3: BN + relu on mutual_raw [32768,192] -> f16 TRANSPOSED mutT[b][192][2048]
// ---------------------------------------------------------------------------
__global__ void k_bn_transpose(const float* __restrict__ x, const float* __restrict__ stats,
                               const float* __restrict__ g, const float* __restrict__ be,
                               _Float16* __restrict__ outT)
{
    int tid = blockIdx.x * blockDim.x + threadIdx.x;   // over 16*192*2048
    int j = tid & (NPTS - 1);
    int c = (tid >> 11) % 192;
    int b = tid / (192 * NPTS);
    float mean = stats[c] * (1.0f / NROWS);
    float var  = stats[192 + c] * (1.0f / NROWS) - mean * mean;
    float sc = g[c] * rsqrtf(var + BN_EPS);
    float sh = be[c] - mean * sc;
    float v = x[((size_t)b * NPTS + j) * 192 + c] * sc + sh;
    outT[((size_t)b * 192 + c) * NPTS + j] = (_Float16)fmaxf(v, 0.0f);
}

// ---------------------------------------------------------------------------
// K4: geometric aggregation with LDS-staged B panel.
// Per 32-j step, the block stages mutT[b][0:192][jb:jb+32] (12KB) in LDS once;
// all 8 waves (i-tiles) consume it -> 8x less L2 traffic. Weight A-matrices are
// generated in-register from xyz (bounded in [0,1], f16-safe), f32 accumulate.
// ---------------------------------------------------------------------------
__global__ void __launch_bounds__(256, 2) k_aggregate(
    const _Float16* __restrict__ mutT, const float* __restrict__ xyz,
    float* __restrict__ ag)
{
    __shared__ __align__(16) _Float16 smem[192 * SROW];

    const int tid  = threadIdx.x;
    const int lane = tid & 31;
    const int wave = tid >> 5;
    const int hi   = lane >> 4;
    const int li   = lane & 15;
    const int b    = blockIdx.y;
    const int ibase = (blockIdx.x * 8 + wave) * 16;
    const int i = ibase + li;

    const float* pi = xyz + ((size_t)b * NPTS + i) * 3;
    const float xi = pi[0], yi = pi[1], zi = pi[2];

    v8f acc0 = {}, acc1 = {};
    float norm = 0.f;
    const float inv_fall = 1.0f / (DRADIUS2 - RADIUS2);
    const _Float16* mb = mutT + (size_t)b * 192 * NPTS;

    // staging: 192 rows x 4 quarters = 768 16B chunks; 3 per thread
    const int idx0 = tid, idx1 = tid + 256, idx2 = tid + 512;

    for (int jb = 0; jb < NPTS; jb += 32) {
        // issue global fetch of this step's B panel (latency hidden by weight VALU)
        uint4 s0 = *(const uint4*)(mb + (size_t)(idx0 >> 2) * NPTS + jb + (idx0 & 3) * 8);
        uint4 s1 = *(const uint4*)(mb + (size_t)(idx1 >> 2) * NPTS + jb + (idx1 & 3) * 8);
        uint4 s2 = *(const uint4*)(mb + (size_t)(idx2 >> 2) * NPTS + jb + (idx2 & 3) * 8);

        // coalesced: lane L fetches xyz of point jb+L, redistributed via ds_bpermute
        const float* pj = xyz + ((size_t)b * NPTS + jb + lane) * 3;
        float xj = pj[0], yj = pj[1], zj = pj[2];

        v16h apx, anx, apy, any_, apz, anz;
#pragma unroll
        for (int h = 0; h < 16; ++h) {
            int sl = h + hi * 16;   // A layout K-map
            float xs = __shfl(xj, sl);
            float ys = __shfl(yj, sl);
            float zs = __shfl(zj, sl);
            float dx = xs - xi, dy = ys - yi, dz = zs - zi;
            float d2 = dx * dx + dy * dy + dz * dz;
            float wd = (d2 <= RADIUS2) ? 1.0f : (DRADIUS2 - d2) * inv_fall;
            wd = (d2 < DRADIUS2 && d2 > 0.0f) ? wd : 0.0f;
            norm += wd;
            float inv = wd * __builtin_amdgcn_rcpf(fmaxf(d2, 1e-12f));
            float px = fmaxf(dx, 0.f), nx = fmaxf(-dx, 0.f);
            float py = fmaxf(dy, 0.f), ny = fmaxf(-dy, 0.f);
            float pz = fmaxf(dz, 0.f), nz = fmaxf(-dz, 0.f);
            apx[h]  = (_Float16)(inv * px * px);
            anx[h]  = (_Float16)(inv * nx * nx);
            apy[h]  = (_Float16)(inv * py * py);
            any_[h] = (_Float16)(inv * ny * ny);
            apz[h]  = (_Float16)(inv * pz * pz);
            anz[h]  = (_Float16)(inv * nz * nz);
        }

        __syncthreads();   // previous step's LDS reads complete
        *(uint4*)(&smem[(idx0 >> 2) * SROW + (idx0 & 3) * 8]) = s0;
        *(uint4*)(&smem[(idx1 >> 2) * SROW + (idx1 & 3) * 8]) = s1;
        *(uint4*)(&smem[(idx2 >> 2) * SROW + (idx2 & 3) * 8]) = s2;
        __syncthreads();   // panel visible to all waves

        // B layout: lane (li,hi) reads row c = g*32 + {0,16}+li, halves [hi*16..+15]
#pragma unroll
        for (int g = 0; g < 6; ++g) {
            const v16h& A = (g == 0) ? apx : (g == 1) ? anx : (g == 2) ? apy
                          : (g == 3) ? any_ : (g == 4) ? apz : anz;
            v16h bt0 = load_v16h(&smem[(g * 32 + li) * SROW + hi * 16]);
            acc0 = wmma_f16(A, bt0, acc0);
            v16h bt1 = load_v16h(&smem[(g * 32 + 16 + li) * SROW + hi * 16]);
            acc1 = wmma_f16(A, bt1, acc1);
        }
    }

    // norm[i]: lane L holds half the j-range for i = L&15; pair-sum across halves
    float ntot = norm + __shfl_xor(norm, 16);
#pragma unroll
    for (int r = 0; r < 8; ++r) {
        int row = r + hi * 8;
        float nr = __shfl(ntot, row);                   // lanes 0..15 hold i = lane
        float invn = __builtin_amdgcn_rcpf(fmaxf(nr, 1e-12f));
        size_t o = ((size_t)b * NPTS + ibase + row) * 32;
        ag[o + li]      = acc0[r] * invn;
        ag[o + 16 + li] = acc1[r] * invn;
    }
}

// ---------------------------------------------------------------------------
// K5: BN + relu on ag [32768,32] -> f16 row-major
// ---------------------------------------------------------------------------
__global__ void k_bn_relu_ag(const float* __restrict__ x, const float* __restrict__ stats,
                             const float* __restrict__ g, const float* __restrict__ be,
                             _Float16* __restrict__ out)
{
    int tid = blockIdx.x * blockDim.x + threadIdx.x;   // 32768*32
    int c = tid & 31;
    float mean = stats[c] * (1.0f / NROWS);
    float var  = stats[32 + c] * (1.0f / NROWS) - mean * mean;
    float sc = g[c] * rsqrtf(var + BN_EPS);
    float sh = be[c] - mean * sc;
    out[tid] = (_Float16)fmaxf(x[tid] * sc + sh, 0.0f);
}

// ---------------------------------------------------------------------------
// K6: pre = agn[32768,32] @ W_ag[32,64] + b_ag + self_feat. K=32 -> one WMMA.
// ---------------------------------------------------------------------------
__global__ void __launch_bounds__(128) k_gemm_ag(
    const _Float16* __restrict__ agn, const float* __restrict__ W_ag,
    const float* __restrict__ b_ag, const float* __restrict__ self_feat,
    float* __restrict__ pre)
{
    const int lane = threadIdx.x & 31;
    const int wave = threadIdx.x >> 5;
    const int hi   = lane >> 4;
    const int col  = lane & 15;
    const int n    = wave * 16 + col;          // 0..63

    v16h bt;
#pragma unroll
    for (int h = 0; h < 16; ++h) {
        int k = (h & 7) + hi * 8 + ((h >= 8) ? 16 : 0);
        bt[h] = (_Float16)W_ag[k * 64 + n];
    }
    float bias = b_ag[n];

    for (int t = 0; t < 16; ++t) {
        int mbase = (blockIdx.x * 16 + t) * 16;
        int row = mbase + col;
        const _Float16* ar = agn + (size_t)row * 32;
        v8h lo  = *(const v8h*)(ar + hi * 8);
        v8h hh  = *(const v8h*)(ar + 16 + hi * 8);
        v16h a;
#pragma unroll
        for (int q = 0; q < 8; ++q) { a[q] = lo[q]; a[q + 8] = hh[q]; }
        v8f c = {};
        c = wmma_f16(a, bt, c);
#pragma unroll
        for (int r = 0; r < 8; ++r) {
            int orow = mbase + r + hi * 8;
            size_t o = (size_t)orow * 64 + n;
            pre[o] = c[r] + bias + self_feat[o];
        }
    }
}

// ---------------------------------------------------------------------------
// K7: final BN + relu -> d_out (f32)
// ---------------------------------------------------------------------------
__global__ void k_final(const float* __restrict__ pre, const float* __restrict__ stats,
                        const float* __restrict__ g, const float* __restrict__ be,
                        float* __restrict__ out)
{
    int tid = blockIdx.x * blockDim.x + threadIdx.x;   // 32768*64
    int c = tid & 63;
    float mean = stats[c] * (1.0f / NROWS);
    float var  = stats[64 + c] * (1.0f / NROWS) - mean * mean;
    float sc = g[c] * rsqrtf(var + BN_EPS);
    float sh = be[c] - mean * sc;
    out[tid] = fmaxf(pre[tid] * sc + sh, 0.0f);
}

extern "C" void kernel_launch(void* const* d_in, const int* in_sizes, int n_in,
                              void* d_out, int out_size, void* d_ws, size_t ws_size,
                              hipStream_t stream)
{
    (void)in_sizes; (void)n_in; (void)out_size; (void)ws_size;
    const float* feat   = (const float*)d_in[0];
    const float* xyz    = (const float*)d_in[1];
    const float* W_feat = (const float*)d_in[2];
    const float* b_feat = (const float*)d_in[3];
    const float* W_byp  = (const float*)d_in[4];
    const float* g_byp  = (const float*)d_in[5];
    const float* be_byp = (const float*)d_in[6];
    const float* g1     = (const float*)d_in[7];
    const float* be1    = (const float*)d_in[8];
    const float* W_ag   = (const float*)d_in[9];
    const float* b_ag   = (const float*)d_in[10];
    const float* g2     = (const float*)d_in[11];
    const float* be2    = (const float*)d_in[12];

    char* ws = (char*)d_ws;
    size_t o = 0;
    float*    self_feat  = (float*)(ws + o);    o += (size_t)NROWS * 64 * 4;
    float*    mutual_raw = (float*)(ws + o);    o += (size_t)NROWS * 192 * 4;
    _Float16* mutT       = (_Float16*)(ws + o); o += (size_t)NB * 192 * NPTS * 2;
    float*    ag_raw     = (float*)(ws + o);    o += (size_t)NROWS * 32 * 4;
    _Float16* agn        = (_Float16*)(ws + o); o += (size_t)NROWS * 32 * 2;
    float*    pre        = (float*)(ws + o);    o += (size_t)NROWS * 64 * 4;
    float*    stats_mut  = (float*)(ws + o);    o += 2 * 192 * 4;
    float*    stats_ag   = (float*)(ws + o);    o += 2 * 32 * 4;
    float*    stats_pre  = (float*)(ws + o);    o += 2 * 64 * 4;

    // zero all BN-stat accumulators (contiguous) — capture-safe
    hipMemsetAsync(stats_mut, 0, (2 * 192 + 2 * 32 + 2 * 64) * sizeof(float), stream);

    k_gemm_in<<<dim3(128, 4), 128, 0, stream>>>(feat, W_feat, b_feat, W_byp,
                                                self_feat, mutual_raw);
    k_stats<<<256, 192, 0, stream>>>(mutual_raw, stats_mut, 192, 128);
    k_bn_transpose<<<(NB * 192 * NPTS) / 256, 256, 0, stream>>>(mutual_raw, stats_mut,
                                                                g_byp, be_byp, mutT);
    k_aggregate<<<dim3(16, 16), 256, 0, stream>>>(mutT, xyz, ag_raw);
    k_stats<<<256, 32, 0, stream>>>(ag_raw, stats_ag, 32, 128);
    k_bn_relu_ag<<<(NROWS * 32) / 256, 256, 0, stream>>>(ag_raw, stats_ag, g1, be1, agn);
    k_gemm_ag<<<128, 128, 0, stream>>>(agn, W_ag, b_ag, self_feat, pre);
    k_stats<<<256, 64, 0, stream>>>(pre, stats_pre, 64, 128);
    k_final<<<(NROWS * 64) / 256, 256, 0, stream>>>(pre, stats_pre, g2, be2, (float*)d_out);
}
